// LinearAttentionRemoteMessageFunction_53910429499703
// MI455X (gfx1250) — matrix operand order repaired and checked by reading, not compiled, not run on hardware
//
#include <hip/hip_runtime.h>

typedef _Float16 v16h __attribute__((ext_vector_type(16)));
typedef _Float16 v8h  __attribute__((ext_vector_type(8)));
typedef __fp16   h2   __attribute__((ext_vector_type(2)));   // cvt_pkrtz result type
typedef float    v8f  __attribute__((ext_vector_type(8)));
typedef unsigned int u32x4 __attribute__((ext_vector_type(4)));
typedef int      i32x4 __attribute__((ext_vector_type(4)));
typedef int      i32x8 __attribute__((ext_vector_type(8)));

#define WMMA_F16(a, b, c) \
  __builtin_amdgcn_wmma_f32_16x16x32_f16(false, (a), false, (b), (short)0, (c), false, false)

#define DN   128
#define HN   8
#define HIDN 256
#define QKN  64
#define VN   64
#define OUTN 128

// packed f16 weight offsets (in f16 elements) inside workspace
#define OFF_K0 0
#define OFF_V0 262144
#define OFF_Q0 524288
#define OFF_K1 786432
#define OFF_V1 917504
#define OFF_Q1 1048576
#define OFF_P0 1179648
#define OFF_P1 1310720
#define F16_TOTAL 1343488
#define KV_ELEMS 32768   // H*64*64
#define SK_ELEMS 512     // H*64

// A-operand (16x32 f16) element -> K index pattern (contiguous: K = e + 16*halfwave, matches B packing)
__device__ __forceinline__ int kpatA(int e, int h) {
  return 16 * h + e;
}

__device__ __forceinline__ v16h load_btile(const _Float16* __restrict__ base, int tile, int lane) {
  return ((const v16h*)(base + tile * 512))[lane];
}

__device__ __forceinline__ float phif(float x) { return x > 0.f ? x + 1.f : __expf(x); }

// ---- Tensor Data Mover: load a 32-row x 128-col f32 tile (row-major, stride 128)
// into LDS at lds_off. rows_valid rows are in-bounds; TDM zero-fills OOB rows.
__device__ __forceinline__ void tdm_load_x_tile(const float* __restrict__ gsrc,
                                                unsigned lds_off, unsigned rows_valid) {
  unsigned long long ga = (unsigned long long)(const void*)gsrc;
  u32x4 g0;
  g0[0] = 1u;                                       // count=1 (valid user descriptor)
  g0[1] = lds_off;                                  // lds_addr
  g0[2] = (unsigned)(ga & 0xffffffffu);             // global_addr[31:0]
  g0[3] = (unsigned)((ga >> 32) & 0x01ffffffu)      // global_addr[56:32]
        | 0x80000000u;                              // type=2 ("image")
  i32x8 g1;
  g1[0] = (int)(2u << 16);                          // data_size = 4 bytes
  g1[1] = (int)(128u << 16);                        // tensor_dim0[15:0] = 128 @ bits 63:48
  g1[2] = (int)((rows_valid & 0xffffu) << 16);      // dim0 hi=0 | tensor_dim1[15:0]
  g1[3] = (int)(((rows_valid >> 16) & 0xffffu)      // tensor_dim1[31:16]
        | (128u << 16));                            // tile_dim0 = 128
  g1[4] = 32;                                       // tile_dim1 = 32 rows, tile_dim2 = 0
  g1[5] = 128;                                      // tensor_dim0_stride = 128 elements
  g1[6] = 0;
  g1[7] = 0;
  i32x4 gz = {0, 0, 0, 0};
#if defined(__clang_major__) && (__clang_major__ >= 23)
  i32x8 gz8 = {0, 0, 0, 0, 0, 0, 0, 0};
  __builtin_amdgcn_tensor_load_to_lds(g0, g1, gz, gz, gz8, 0);
#else
  __builtin_amdgcn_tensor_load_to_lds(g0, g1, gz, gz, 0);
#endif
}

// Fused per-head 2-layer MLP: x(32x128) -> relu(x@W0+b0)(32x256) -> @W1 (32x64).
// Second-layer bias applied by caller. swpw = per-wave 16x32 f16 LDS scratch
// (column-major [hid-col][point]) for the C->A layout round trip.
__device__ __forceinline__ void mlp_head(
    const v16h (&ax)[2][4],
    const _Float16* __restrict__ w0, const _Float16* __restrict__ w1,
    const float* __restrict__ b0,
    _Float16* __restrict__ swpw, int lane, v8f (&out)[2][4])
{
  const int col = lane & 15, half = lane >> 4;
  #pragma unroll
  for (int mt = 0; mt < 2; ++mt)
    #pragma unroll
    for (int nt = 0; nt < 4; ++nt) out[mt][nt] = {};

  for (int hc = 0; hc < 8; ++hc) {           // 32-wide chunks of hidden dim
    #pragma unroll
    for (int mt = 0; mt < 2; ++mt) {
      #pragma unroll
      for (int nt2 = 0; nt2 < 2; ++nt2) {
        v8f t = {};
        #pragma unroll
        for (int kt = 0; kt < 4; ++kt) {     // K = 128 input dim
          v16h b = load_btile(w0, kt * 16 + hc * 2 + nt2, lane);
          t = WMMA_F16(ax[mt][kt], b, t);
        }
        float bias = b0[(hc * 2 + nt2) * 16 + col];
        union { v8h v; h2 p[4]; } s;
        #pragma unroll
        for (int r2 = 0; r2 < 4; ++r2) {
          float a = t[2 * r2] + bias, b2 = t[2 * r2 + 1] + bias;
          a = a > 0.f ? a : 0.f;
          b2 = b2 > 0.f ? b2 : 0.f;
          s.p[r2] = __builtin_amdgcn_cvt_pkrtz(a, b2);
        }
        *(v8h*)&swpw[(nt2 * 16 + col) * 16 + half * 8] = s.v;  // packed b128 store
      }
      v16h a2;  // hidden chunk as layer-2 A operand (transposed read)
      #pragma unroll
      for (int e = 0; e < 16; ++e) a2[e] = swpw[kpatA(e, half) * 16 + col];
      #pragma unroll
      for (int nt = 0; nt < 4; ++nt) {       // 64 output cols
        v16h b = load_btile(w1, hc * 4 + nt, lane);
        out[mt][nt] = WMMA_F16(a2, b, out[mt][nt]);
      }
    }
  }
}

// ---- weight prep: fp32 row-major (heads,K,Nc) -> f16 B-operand tiles ----
__global__ void pack_b_kernel(const float* __restrict__ src, _Float16* __restrict__ dst,
                              int K, int Nc, int total) {
  int idx = blockIdx.x * blockDim.x + threadIdx.x;
  if (idx >= total) return;
  int e = idx & 15, lane = (idx >> 4) & 31, t = idx >> 9;
  int ntiles = Nc >> 4, ktiles = K >> 5;
  int nt = t % ntiles; t /= ntiles;
  int kt = t % ktiles; int hh = t / ktiles;
  int k  = kt * 32 + (lane >> 4) * 16 + e;   // B layout: K = e + 16*halfwave
  int nc = nt * 16 + (lane & 15);
  dst[idx] = (_Float16)src[(hh * K + k) * Nc + nc];
}

__global__ void zero_kernel(float* p, int count) {
  int i = blockIdx.x * blockDim.x + threadIdx.x;
  if (i < count) p[i] = 0.f;
}

// ---- pass 1: K/V MLPs + global kv / sum_k reduction ----
__global__ __launch_bounds__(256) void pass1_kernel(
    const float* __restrict__ x, const _Float16* __restrict__ wp,
    const float* __restrict__ bk0, const float* __restrict__ bk1,
    const float* __restrict__ bv0, const float* __restrict__ bv1,
    float* __restrict__ kvg, float* __restrict__ skg, int n)
{
  __shared__ __align__(16) float    xs[32 * DN];       // 16 KB : x tile (f32, TDM-filled)
  __shared__ __align__(16) _Float16 vs[HN][32 * VN];   // 32 KB : col-major [vcol][point]
  __shared__ __align__(16) _Float16 swp[HN][512];      //  8 KB : per-wave MLP scratch

  const int tid = threadIdx.x, w = tid >> 5, lane = tid & 31;
  const int col = lane & 15, half = lane >> 4;
  const unsigned xs_lds = (unsigned)(unsigned long long)(void*)xs;

  v8f kvacc[4][4];
  #pragma unroll
  for (int a = 0; a < 4; ++a)
    #pragma unroll
    for (int b = 0; b < 4; ++b) kvacc[a][b] = {};
  float skacc[4] = {0.f, 0.f, 0.f, 0.f};

  for (int it = 0; it < 8; ++it) {        // 8 x 32 = 256 rows per block
    const int i0 = (blockIdx.x * 8 + it) * 32;
    if (w == 0) {                          // one TDM descriptor per tile
      long long rem = (long long)n - (long long)i0;
      if (rem < 0) rem = 0;
      tdm_load_x_tile(x + (size_t)i0 * DN, xs_lds, (unsigned)rem);
      __builtin_amdgcn_s_wait_tensorcnt(0);
    }
    __syncthreads();

    if (it + 1 < 8) {                      // warm L2 for the next tile
      const float* nx = x + ((size_t)i0 + 32) * DN;
      __builtin_prefetch(nx + (size_t)(tid >> 3) * DN + (size_t)(tid & 7) * 16, 0, 0);
    }

    v16h ax[2][4];                         // x as A operands, K = 128 (pk-converted)
    #pragma unroll
    for (int mt = 0; mt < 2; ++mt)
      #pragma unroll
      for (int kt = 0; kt < 4; ++kt) {
        union { v16h v; h2 p[8]; } u;
        #pragma unroll
        for (int e2 = 0; e2 < 8; ++e2) {
          int idx = (mt * 16 + col) * DN + kt * 32 + kpatA(2 * e2, half);
          float2 f = *(const float2*)&xs[idx];
          u.p[e2] = __builtin_amdgcn_cvt_pkrtz(f.x, f.y);
        }
        ax[mt][kt] = u.v;
      }

    // ---- K branch ----
    v8f kk[2][4];
    mlp_head(ax, wp + OFF_K0 + w * 32768, wp + OFF_K1 + w * 16384,
             bk0 + w * HIDN, swp[w], lane, kk);
    #pragma unroll
    for (int mt = 0; mt < 2; ++mt)
      #pragma unroll
      for (int nt = 0; nt < 4; ++nt) {
        float bias = bk1[w * QKN + nt * 16 + col];
        #pragma unroll
        for (int r = 0; r < 8; ++r) {
          int row = i0 + mt * 16 + r + half * 8;
          kk[mt][nt][r] = (row < n) ? phif(kk[mt][nt][r] + bias) : 0.f;
        }
      }
    #pragma unroll
    for (int nt = 0; nt < 4; ++nt) {
      float s = 0.f;
      #pragma unroll
      for (int r = 0; r < 8; ++r) s += kk[0][nt][r] + kk[1][nt][r];
      skacc[nt] += s;
    }

    // ---- V branch ----
    v8f vv[2][4];
    mlp_head(ax, wp + OFF_V0 + w * 32768, wp + OFF_V1 + w * 16384,
             bv0 + w * HIDN, swp[w], lane, vv);
    #pragma unroll
    for (int mt = 0; mt < 2; ++mt)
      #pragma unroll
      for (int nt = 0; nt < 4; ++nt) {
        float bias = bv1[w * VN + nt * 16 + col];
        union { v8h v; h2 p[4]; } s;
        #pragma unroll
        for (int r2 = 0; r2 < 4; ++r2) {
          int row0 = i0 + mt * 16 + 2 * r2 + half * 8;
          float a  = (row0 < n)     ? (vv[mt][nt][2 * r2]     + bias) : 0.f;
          float b2 = (row0 + 1 < n) ? (vv[mt][nt][2 * r2 + 1] + bias) : 0.f;
          s.p[r2] = __builtin_amdgcn_cvt_pkrtz(a, b2);
        }
        // col-major staging: [vcol][point] -> one packed b128 store
        *(v8h*)&vs[w][(nt * 16 + col) * 32 + mt * 16 + half * 8] = s.v;
      }

    // ---- kv += kk^T (64x32) @ v (32x64) ----
    // A operand of kk^T comes straight from the C tiles (in-lane transpose).
    #pragma unroll
    for (int kt = 0; kt < 4; ++kt) {
      union { v16h v; h2 p[8]; } ua;
      #pragma unroll
      for (int e2 = 0; e2 < 4; ++e2)
        ua.p[e2] = __builtin_amdgcn_cvt_pkrtz(kk[0][kt][2 * e2], kk[0][kt][2 * e2 + 1]);
      #pragma unroll
      for (int e2 = 4; e2 < 8; ++e2)
        ua.p[e2] = __builtin_amdgcn_cvt_pkrtz(kk[1][kt][2 * e2 - 8], kk[1][kt][2 * e2 - 7]);
      #pragma unroll
      for (int vt = 0; vt < 4; ++vt) {
        v8h bl = *(const v8h*)&vs[w][(vt * 16 + col) * 32 + half * 8];
        v8h bh = *(const v8h*)&vs[w][(vt * 16 + col) * 32 + 16 + half * 8];
        v16h bv = __builtin_shufflevector(bl, bh, 0, 1, 2, 3, 4, 5, 6, 7, 8, 9, 10, 11, 12, 13, 14, 15); // 2x ds_load_b128
        kvacc[kt][vt] = WMMA_F16(ua.v, bv, kvacc[kt][vt]);
      }
    }
    __syncthreads();
  }

  // flush partial reductions (f32 atomics into a 128 KB L2-resident region)
  #pragma unroll
  for (int kt = 0; kt < 4; ++kt) {
    atomicAdd(&skg[w * QKN + kt * 16 + col], skacc[kt]);
    #pragma unroll
    for (int vt = 0; vt < 4; ++vt)
      #pragma unroll
      for (int r = 0; r < 8; ++r)
        atomicAdd(&kvg[w * (QKN * VN) + (kt * 16 + r + half * 8) * VN + vt * 16 + col],
                  kvacc[kt][vt][r]);
  }
}

// ---- pass 2: Q MLP + attention readout + psi MLP ----
__global__ __launch_bounds__(256) void pass2_kernel(
    const float* __restrict__ x, const float* __restrict__ maskp,
    const _Float16* __restrict__ wp,
    const float* __restrict__ bq0, const float* __restrict__ bq1,
    const float* __restrict__ bp0, const float* __restrict__ bp1,
    const float* __restrict__ kvg, const float* __restrict__ skg,
    float* __restrict__ out, int n)
{
  __shared__ __align__(16) unsigned char ubuf[16384]; // xs f32 (16KB) overlaid with h1s (8KB)
  __shared__ __align__(16) _Float16 heads_s[16 * 512];// 16 KB : concat heads, one m-tile
  __shared__ __align__(16) _Float16 swp[HN][1024];    // 16 KB : per-wave scratch
  float*    xs  = (float*)ubuf;
  _Float16* h1s = (_Float16*)ubuf;

  const int tid = threadIdx.x, w = tid >> 5, lane = tid & 31;
  const int col = lane & 15, half = lane >> 4;
  const int i0 = blockIdx.x * 32;

  if (w == 0) {
    long long rem = (long long)n - (long long)i0;
    if (rem < 0) rem = 0;
    tdm_load_x_tile(x + (size_t)i0 * DN, (unsigned)(unsigned long long)(void*)xs, (unsigned)rem);
    __builtin_amdgcn_s_wait_tensorcnt(0);
  }
  __syncthreads();

  v16h ax[2][4];
  #pragma unroll
  for (int mt = 0; mt < 2; ++mt)
    #pragma unroll
    for (int kt = 0; kt < 4; ++kt) {
      union { v16h v; h2 p[8]; } u;
      #pragma unroll
      for (int e2 = 0; e2 < 8; ++e2) {
        int idx = (mt * 16 + col) * DN + kt * 32 + kpatA(2 * e2, half);
        float2 f = *(const float2*)&xs[idx];
        u.p[e2] = __builtin_amdgcn_cvt_pkrtz(f.x, f.y);
      }
      ax[mt][kt] = u.v;
    }

  v8f qk[2][4];
  mlp_head(ax, wp + OFF_Q0 + w * 32768, wp + OFF_Q1 + w * 16384,
           bq0 + w * HIDN, swp[w], lane, qk);
  #pragma unroll
  for (int mt = 0; mt < 2; ++mt)
    #pragma unroll
    for (int nt = 0; nt < 4; ++nt) {
      float bias = bq1[w * QKN + nt * 16 + col];
      #pragma unroll
      for (int r = 0; r < 8; ++r) qk[mt][nt][r] = phif(qk[mt][nt][r] + bias);
    }

  for (int mt = 0; mt < 2; ++mt) {
    // stage qk m-tile col-major [k][point] with packed stores
    #pragma unroll
    for (int nt = 0; nt < 4; ++nt) {
      union { v8h v; h2 p[4]; } s;
      #pragma unroll
      for (int r2 = 0; r2 < 4; ++r2)
        s.p[r2] = __builtin_amdgcn_cvt_pkrtz(qk[mt][nt][2 * r2], qk[mt][nt][2 * r2 + 1]);
      *(v8h*)&swp[w][(nt * 16 + col) * 16 + half * 8] = s.v;
    }

    // denominator: lane&15 owns point p = col
    float den = 0.f;
    for (int k = 0; k < QKN; ++k)
      den += (float)swp[w][k * 16 + col] * skg[w * QKN + k];
    if (den == 0.f) den = 1e-6f;

    // num = qk (16x64) @ kv (64x64)
    v8f nacc[4];
    #pragma unroll
    for (int vt = 0; vt < 4; ++vt) nacc[vt] = {};
    #pragma unroll
    for (int kc = 0; kc < 2; ++kc) {
      v16h aq;
      #pragma unroll
      for (int e = 0; e < 16; ++e)
        aq[e] = swp[w][(kc * 32 + kpatA(e, half)) * 16 + col];
      #pragma unroll
      for (int vt = 0; vt < 4; ++vt) {
        v16h bkv;
        #pragma unroll
        for (int e = 0; e < 16; ++e)
          bkv[e] = (_Float16)kvg[w * (QKN * VN) + (kc * 32 + e + half * 16) * VN + vt * 16 + col];
        nacc[vt] = WMMA_F16(aq, bkv, nacc[vt]);
      }
    }
    #pragma unroll
    for (int vt = 0; vt < 4; ++vt)
      #pragma unroll
      for (int r = 0; r < 8; ++r) {
        float d = __shfl(den, r + half * 8, 32);
        heads_s[(r + half * 8) * 512 + w * VN + vt * 16 + col] = (_Float16)(nacc[vt][r] / d);
      }
    __syncthreads();

    // psi layer 1: wave w owns cols [w*32, w*32+32)
    v8f p1[2]; p1[0] = {}; p1[1] = {};
    for (int kc = 0; kc < 16; ++kc) {
      v16h a;
      #pragma unroll
      for (int e = 0; e < 16; ++e) a[e] = heads_s[col * 512 + kc * 32 + kpatA(e, half)];
      #pragma unroll
      for (int nt2 = 0; nt2 < 2; ++nt2) {
        v16h b = load_btile(wp + OFF_P0, kc * 16 + w * 2 + nt2, lane);
        p1[nt2] = WMMA_F16(a, b, p1[nt2]);
      }
    }
    #pragma unroll
    for (int nt2 = 0; nt2 < 2; ++nt2) {
      int cp = w * 32 + nt2 * 16 + col;
      float bias = bp0[cp];
      #pragma unroll
      for (int r = 0; r < 8; ++r) {
        float v = p1[nt2][r] + bias;
        v = v > 0.f ? v : 0.f;
        h1s[(r + half * 8) * 256 + cp] = (_Float16)v;
      }
    }
    __syncthreads();

    // psi layer 2: wave w owns out cols [w*16, w*16+16)
    v8f po = {};
    #pragma unroll
    for (int kc = 0; kc < 8; ++kc) {
      v16h a;
      #pragma unroll
      for (int e = 0; e < 16; ++e) a[e] = h1s[col * 256 + kc * 32 + kpatA(e, half)];
      v16h b = load_btile(wp + OFF_P1, kc * 8 + w, lane);
      po = WMMA_F16(a, b, po);
    }
    float bias = bp1[w * 16 + col];
    #pragma unroll
    for (int r = 0; r < 8; ++r) {
      int row = i0 + mt * 16 + r + half * 8;
      if (row < n)
        out[row * OUTN + w * 16 + col] = (po[r] + bias) * maskp[row];
    }
    __syncthreads();
  }
}

extern "C" void kernel_launch(void* const* d_in, const int* in_sizes, int n_in,
                              void* d_out, int out_size, void* d_ws, size_t ws_size,
                              hipStream_t stream) {
  const float* x   = (const float*)d_in[0];
  const float* msk = (const float*)d_in[1];
  const float* Wq0 = (const float*)d_in[2];  const float* bq0 = (const float*)d_in[3];
  const float* Wq1 = (const float*)d_in[4];  const float* bq1 = (const float*)d_in[5];
  const float* Wk0 = (const float*)d_in[6];  const float* bk0 = (const float*)d_in[7];
  const float* Wk1 = (const float*)d_in[8];  const float* bk1 = (const float*)d_in[9];
  const float* Wv0 = (const float*)d_in[10]; const float* bv0 = (const float*)d_in[11];
  const float* Wv1 = (const float*)d_in[12]; const float* bv1 = (const float*)d_in[13];
  const float* Wp0 = (const float*)d_in[14]; const float* bp0 = (const float*)d_in[15];
  const float* Wp1 = (const float*)d_in[16]; const float* bp1 = (const float*)d_in[17];
  float* out = (float*)d_out;
  const int n = in_sizes[0] / DN;

  _Float16* wp = (_Float16*)d_ws;
  float* kvg = (float*)((char*)d_ws + (size_t)F16_TOTAL * sizeof(_Float16));
  float* skg = kvg + KV_ELEMS;

  // weight packing (f32 -> f16 B-tile layout)
  {
    int t;
    t = 8 * 4 * 16 * 512;
    pack_b_kernel<<<(t + 255) / 256, 256, 0, stream>>>(Wk0, wp + OFF_K0, 128, 256, t);
    pack_b_kernel<<<(t + 255) / 256, 256, 0, stream>>>(Wv0, wp + OFF_V0, 128, 256, t);
    pack_b_kernel<<<(t + 255) / 256, 256, 0, stream>>>(Wq0, wp + OFF_Q0, 128, 256, t);
    t = 8 * 8 * 4 * 512;
    pack_b_kernel<<<(t + 255) / 256, 256, 0, stream>>>(Wk1, wp + OFF_K1, 256, 64, t);
    pack_b_kernel<<<(t + 255) / 256, 256, 0, stream>>>(Wv1, wp + OFF_V1, 256, 64, t);
    pack_b_kernel<<<(t + 255) / 256, 256, 0, stream>>>(Wq1, wp + OFF_Q1, 256, 64, t);
    t = 16 * 16 * 512;
    pack_b_kernel<<<(t + 255) / 256, 256, 0, stream>>>(Wp0, wp + OFF_P0, 512, 256, t);
    t = 8 * 8 * 512;
    pack_b_kernel<<<(t + 255) / 256, 256, 0, stream>>>(Wp1, wp + OFF_P1, 256, 128, t);
  }
  zero_kernel<<<(KV_ELEMS + SK_ELEMS + 255) / 256, 256, 0, stream>>>(kvg, KV_ELEMS + SK_ELEMS);

  const int blocks1 = (n + 255) / 256;
  pass1_kernel<<<blocks1, 256, 0, stream>>>(x, wp, bk0, bk1, bv0, bv1, kvg, skg, n);
  const int blocks2 = (n + 31) / 32;
  pass2_kernel<<<blocks2, 256, 0, stream>>>(x, msk, wp, bq0, bq1, bp0, bp1, kvg, skg, out, n);
}
